// Conv2d_res_bit_baens_61942018343229
// MI455X (gfx1250) — matrix operand order repaired and checked by reading, not compiled, not run on hardware
//
#include <hip/hip_runtime.h>
#include <hip/hip_bf16.h>

typedef __attribute__((ext_vector_type(16))) _Float16 v16h;
typedef __attribute__((ext_vector_type(8)))  _Float16 v8h;
typedef __attribute__((ext_vector_type(8)))  float    v8f;

#define NENS 5
#define IN_CH 128
#define OUT_CH 128
#define KTOT (IN_CH * 9)        // 1152 GEMM-K
#define WPERG (OUT_CH * KTOT)   // 147456 halfs per ensemble member
#define IMG_ELEMS (IN_CH * 32 * 32)

// ---------------------------------------------------------------------------
// 1) global min / max of U  (737280 elems = 1024 threads * 720)
// ---------------------------------------------------------------------------
__global__ __launch_bounds__(1024) void minmax_kernel(const float* __restrict__ U,
                                                      int n, float* __restrict__ mm) {
    __shared__ float smin[1024];
    __shared__ float smax[1024];
    int t = threadIdx.x;
    float mn = 3.4e38f, mx = -3.4e38f;
    for (int i = t; i < n; i += 1024) {
        float v = U[i];
        mn = fminf(mn, v);
        mx = fmaxf(mx, v);
    }
    smin[t] = mn; smax[t] = mx;
    __syncthreads();
    for (int s = 512; s > 0; s >>= 1) {
        if (t < s) {
            smin[t] = fminf(smin[t], smin[t + s]);
            smax[t] = fmaxf(smax[t], smax[t + s]);
        }
        __syncthreads();
    }
    if (t == 0) { mm[0] = smin[0]; mm[1] = smax[0]; }
}

// ---------------------------------------------------------------------------
// 2) residual-bit quantization:  w = v1 + v2  -> f16, layout [g][oc][k]
// ---------------------------------------------------------------------------
__global__ __launch_bounds__(256) void quant_kernel(const float* __restrict__ U,
                                                    const float* __restrict__ mm,
                                                    _Float16* __restrict__ wq, int n) {
    int i = blockIdx.x * blockDim.x + threadIdx.x;
    if (i >= n) return;
    float alpha = mm[0], beta = mm[1];
    float s1 = (beta - alpha) * (1.0f / 3.0f);
    float s2 = s1 * (1.0f / 5.0f);
    float u  = U[i];
    float v1 = s1 * floorf(u / s1);
    float v2 = s2 * floorf((u - v1) / s2);
    wq[i] = (_Float16)(v1 + v2);
}

// ---------------------------------------------------------------------------
// 3) implicit-GEMM conv, one image per blockIdx.y, 128-pixel tile per blockIdx.x
//    M = 128 oc, N = 128 px, K = 1152, WMMA f32 16x16x32 f16
//    A-tile staged via async direct-to-LDS copies (ASYNCcnt path)
// ---------------------------------------------------------------------------
__global__ __launch_bounds__(256) void conv_kernel(const float* __restrict__ x,
                                                   const _Float16* __restrict__ wq,
                                                   float* __restrict__ out) {
    __shared__ _Float16 Ash[128 * 32];   // [oc][k]  8 KB
    __shared__ _Float16 Bsh[128 * 32];   // [px][k]  8 KB (im2col, K contiguous)

    const int b      = blockIdx.y;            // original image index
    const int g      = b % NENS;              // ensemble member / conv group
    const int pxbase = blockIdx.x * 128;      // 4 output rows of 32
    const int tid    = threadIdx.x;
    const int lane   = tid & 31;
    const int wave   = tid >> 5;              // 8 waves -> 8 strips of 16 oc
    const int lm     = lane & 15;
    const int hi     = lane >> 4;             // lane-group select (0/1)

    const float*    ximg = x  + (size_t)b * IMG_ELEMS;
    const _Float16* wg   = wq + (size_t)g * WPERG;

    v8f zero = {};
    v8f acc[8];
#pragma unroll
    for (int n = 0; n < 8; ++n) acc[n] = zero;

    const int fo = tid >> 1;         // 0..127 : oc row (A) / pixel row (B)
    const int fh = (tid & 1) * 16;   // which 16-half chunk of the 32-k row

    const int pg = pxbase + fo;
    const int py = pg >> 5;
    const int px = pg & 31;

    // incremental decomposition of k = ic*9 + r  (r = kh*3+kw), k0 = fh
    int ic0 = (fh == 0) ? 0 : 1;
    int r0  = (fh == 0) ? 0 : 7;

    // per-thread LDS/global addresses for the async A-tile copy (32 B each)
    const unsigned a_lds = (unsigned)(uintptr_t)&Ash[fo * 32 + fh];

    for (int ks = 0; ks < 36; ++ks) {
        const int kbase = ks * 32;
        __syncthreads();

        // ---- stage A tile: 128 oc x 32 k via async global->LDS (no VGPR bounce)
        {
            unsigned long long ga =
                (unsigned long long)(uintptr_t)(wg + (size_t)fo * KTOT + kbase + fh);
            asm volatile("global_load_async_to_lds_b128 %0, %1, off"
                         :: "v"(a_lds), "v"(ga) : "memory");
            asm volatile("global_load_async_to_lds_b128 %0, %1, off offset:16"
                         :: "v"(a_lds), "v"(ga) : "memory");
            if (ks < 35)
                __builtin_prefetch(wg + (size_t)fo * KTOT + kbase + 32 + fh, 0, 0);
        }
        // ---- stage B tile: im2col gather, 128 px x 32 k, zero-padded borders
        {
            int ic = ic0, r = r0;
#pragma unroll
            for (int j = 0; j < 16; ++j) {
                int kh = (r >= 6) ? 2 : ((r >= 3) ? 1 : 0);
                int kw = r - kh * 3;
                int iy = py + kh - 1;
                int ix = px + kw - 1;
                float v = 0.0f;
                if ((unsigned)iy < 32u && (unsigned)ix < 32u)
                    v = ximg[(ic * 32 + iy) * 32 + ix];
                Bsh[fo * 32 + fh + j] = (_Float16)v;
                if (++r == 9) { r = 0; ++ic; }
            }
        }
        // advance (ic0, r0) by 32 k-positions for the next k-step
        r0 += 5; ic0 += 3;
        if (r0 >= 9) { r0 -= 9; ++ic0; }

        // wait for this wave's async copies, then block-wide barrier
        asm volatile("s_wait_asynccnt 0x0" ::: "memory");
        __syncthreads();

        // ---- A fragment for this wave's 16-oc strip (ISA 16-bit A layout)
        v16h af;
        {
            const int kq = hi ? 8 : 0;
            v8h a0 = *(const v8h*)&Ash[(wave * 16 + lm) * 32 + kq];        // K kq..kq+7
            v8h a1 = *(const v8h*)&Ash[(wave * 16 + lm) * 32 + kq + 16];   // K kq+16..kq+23
#pragma unroll
            for (int i = 0; i < 8; ++i) { af[i] = a0[i]; af[i + 8] = a1[i]; }
        }

        // ---- 8 pixel-column tiles of 16: B fragment + WMMA accumulate
        const int bq = hi ? 8 : 0;
#pragma unroll
        for (int n = 0; n < 8; ++n) {
            v8h b0 = *(const v8h*)&Bsh[(n * 16 + lm) * 32 + bq];       // K bq..bq+7
            v8h b1 = *(const v8h*)&Bsh[(n * 16 + lm) * 32 + bq + 16];  // K bq+16..bq+23
            v16h bf;
#pragma unroll
            for (int i = 0; i < 8; ++i) { bf[i] = b0[i]; bf[i + 8] = b1[i]; }
            acc[n] = __builtin_amdgcn_wmma_f32_16x16x32_f16(
                false, af, false, bf, (short)0, acc[n], false, false);
        }
    }

    // ---- store D: VGPR r holds M = r + 8*hi, lane lm holds N (ISA C/D layout)
    float* oimg = out + (size_t)b * (OUT_CH * 1024);
#pragma unroll
    for (int n = 0; n < 8; ++n) {
        int pcol = pxbase + n * 16 + lm;
#pragma unroll
        for (int r = 0; r < 8; ++r) {
            int oc = wave * 16 + r + hi * 8;
            oimg[oc * 1024 + pcol] = acc[n][r];
        }
    }
}

// ---------------------------------------------------------------------------
extern "C" void kernel_launch(void* const* d_in, const int* in_sizes, int n_in,
                              void* d_out, int out_size, void* d_ws, size_t ws_size,
                              hipStream_t stream) {
    const float* x = (const float*)d_in[0];   // (160,128,32,32) f32
    const float* U = (const float*)d_in[1];   // (5,147456) f32
    float* out = (float*)d_out;               // (160,128,32,32) f32

    float*    mm  = (float*)d_ws;                         // [min, max]
    _Float16* wqp = (_Float16*)((char*)d_ws + 64);        // 5*147456 f16 quantized weights

    const int nU = in_sizes[1];               // 737280

    minmax_kernel<<<1, 1024, 0, stream>>>(U, nU, mm);
    quant_kernel<<<(nU + 255) / 256, 256, 0, stream>>>(U, mm, wqp, nU);

    dim3 grid(8, 160);                         // 8 pixel tiles x 160 images
    conv_kernel<<<grid, 256, 0, stream>>>(x, wqp, out);
}
